// AttnProcessor_16028817949036
// MI455X (gfx1250) — matrix operand order repaired and checked by reading, not compiled, not run on hardware
//
#include <hip/hip_runtime.h>
#include <hip/hip_bf16.h>

typedef __attribute__((ext_vector_type(16))) _Float16 v16h;
typedef __attribute__((ext_vector_type(8)))  _Float16 v8h;
typedef __attribute__((ext_vector_type(8)))  float    v8f;
typedef __attribute__((ext_vector_type(4)))  float    f4;
typedef __attribute__((ext_vector_type(4)))  unsigned int u32x4;
typedef __attribute__((ext_vector_type(8)))  int      i32x8;
typedef __attribute__((ext_vector_type(4)))  int      i32x4;

#define B_  2
#define S_  4096
#define C_  512
#define H_  8
#define D_  64
#define M_  (B_*S_)   // 8192 total rows

#if defined(__AMDGCN__) && __has_builtin(__builtin_amdgcn_tensor_load_to_lds) && \
    __has_builtin(__builtin_amdgcn_s_wait_tensorcnt)
#define HAVE_TDM 1
#else
#define HAVE_TDM 0
#endif

// ---------------------------------------------------------------------------
// WMMA helpers (wave32, 16x16x32 f16 -> f32)
// ---------------------------------------------------------------------------
__device__ __forceinline__ v8f wmma16(v16h a, v16h b, v8f c) {
  // 8 args: (neg_a, A, neg_b, B, c_mod, C, reuse_a, reuse_b)
  return __builtin_amdgcn_wmma_f32_16x16x32_f16(false, a, false, b, (short)0, c,
                                                false, false);
}

// A fragment: 16(M) x 32(K), row-major source with leading dim `ld` (halfs).
// row = lane&15; lanes 0-15 hold K 0-7,16-23; lanes 16-31 hold K 8-15,24-31.
__device__ __forceinline__ v16h frag_a(const _Float16* src, int ld) {
  const int lane = threadIdx.x & 31;
  const int row  = lane & 15;
  const int kb   = (lane >> 4) << 3;          // 0 or 8
  const _Float16* p = src + row * ld + kb;
  v8h lo = *(const v8h*)(p);
  v8h hi = *(const v8h*)(p + 16);
  v16h a;
#pragma unroll
  for (int i = 0; i < 8; ++i) { a[i] = lo[i]; a[i + 8] = hi[i]; }
  return a;
}

// B fragment: 32(K) x 16(N), source stored N-major ([n][k], leading dim ld).
// col n = lane&15; lanes 0-15 hold K 0-7,16-23; lanes 16-31 hold K 8-15,24-31.
__device__ __forceinline__ v16h frag_b(const _Float16* src, int ld) {
  const int lane = threadIdx.x & 31;
  const int n    = lane & 15;
  const int kb   = (lane >> 4) << 3;          // 0 or 8
  const _Float16* p = src + n * ld + kb;
  v8h lo = *(const v8h*)(p);
  v8h hi = *(const v8h*)(p + 16);
  v16h b;
#pragma unroll
  for (int i = 0; i < 8; ++i) { b[i] = lo[i]; b[i + 8] = hi[i]; }
  return b;
}

__device__ __forceinline__ v8f vzero8() {
  v8f z = {0.f, 0.f, 0.f, 0.f, 0.f, 0.f, 0.f, 0.f};
  return z;
}

#if HAVE_TDM
// ---------------------------------------------------------------------------
// Tensor Data Mover: 2D tile load global -> LDS with per-row LDS padding.
//  tile_w8 / stride8 : row width / row stride in 8-byte units (data_size=3)
//  pad_iv_code       : pad after (1<<code) 8-byte units stored
//  pad_amt_code      : pad amount = (code+1) DWORDs
// D# layout per CDNA5 ISA (2D, groups 2/3 unused).
// ---------------------------------------------------------------------------
__device__ __forceinline__ void tdm_load_2d(void* lds_dst, const void* gsrc,
                                            unsigned tile_w8, unsigned rows,
                                            unsigned stride8,
                                            unsigned pad_iv_code,
                                            unsigned pad_amt_code) {
  const unsigned lds_off = (unsigned)(uintptr_t)lds_dst;   // addr[31:0] = LDS offset
  const unsigned long long ga = (unsigned long long)(uintptr_t)gsrc;
  u32x4 g0;
  g0[0] = 1u;                                   // count=1 (valid), no gather
  g0[1] = lds_off;                              // lds_addr  (bits 63:32)
  g0[2] = (unsigned)(ga & 0xffffffffu);         // global_addr[31:0]
  g0[3] = (unsigned)((ga >> 32) & 0x01ffffffu)  // global_addr[56:32]
        | (2u << 30);                           // type=2 ("image")
  i32x8 g1;
  g1[0] = (int)((3u << 16)                      // data_size = 8B
              | (1u << 20)                      // pad_enable
              | (pad_iv_code << 22)
              | (pad_amt_code << 25));
  g1[1] = (int)(tile_w8 << 16);                 // tensor_dim0[15:0] @ 63:48
  g1[2] = (int)((tile_w8 >> 16) | (rows << 16));// tensor_dim0 hi | tensor_dim1 lo
  g1[3] = (int)((rows >> 16) | (tile_w8 << 16));// tensor_dim1 hi | tile_dim0
  g1[4] = (int)(rows & 0xffffu);                // tile_dim1 (tile_dim2 = 0)
  g1[5] = (int)stride8;                         // tensor_dim0_stride[31:0]
  g1[6] = 0;                                    // stride0 hi | stride1 lo
  g1[7] = 0;                                    // stride1 hi
  i32x4 gz = {0, 0, 0, 0};
#if __clang_major__ >= 23
  i32x8 gz8 = {0, 0, 0, 0, 0, 0, 0, 0};
  __builtin_amdgcn_tensor_load_to_lds(g0, g1, gz, gz, gz8, 0);
#else
  __builtin_amdgcn_tensor_load_to_lds(g0, g1, gz, gz, 0);
#endif
}
#endif  // HAVE_TDM

// ---------------------------------------------------------------------------
// Kernel 1: fused QKV projection.  X[8192,512]f32 @ W{q,k,v}^T -> f16
// Q pre-scaled by D^-1/2 = 0.125.  Output layout [B,H,S,D].
// ---------------------------------------------------------------------------
__global__ __launch_bounds__(256) void k_qkv(
    const float* __restrict__ X,
    const float* __restrict__ Wq, const float* __restrict__ Wk,
    const float* __restrict__ Wv,
    _Float16* __restrict__ Qh, _Float16* __restrict__ Kh,
    _Float16* __restrict__ Vh)
{
  __shared__ _Float16 sA[128][40];        // X tile, f16, padded stride
  __shared__ _Float16 sB[3][64][40];      // Wq/Wk/Wv tiles, [n][k] layout

  const int tid  = threadIdx.x;
  const int wave = tid >> 5;
  const int lane = tid & 31;
  const int m0   = blockIdx.x * 128;
  const int n0   = blockIdx.y * 64;

  v8f accq[4], acck[4], accv[4];
#pragma unroll
  for (int nt = 0; nt < 4; ++nt) { accq[nt] = vzero8(); acck[nt] = vzero8(); accv[nt] = vzero8(); }

  const float* Wp[3] = {Wq, Wk, Wv};

  for (int kk = 0; kk < C_; kk += 32) {
    // --- stage A: 128 rows x 32 k, f32 -> f16 (2 threads per row)
    {
      const int row = tid >> 1;
      const int c0  = (tid & 1) * 16;
      const float* src = X + (size_t)(m0 + row) * C_ + kk + c0;
#pragma unroll
      for (int j = 0; j < 2; ++j) {
        f4 f0 = ((const f4*)src)[2 * j];
        f4 f1 = ((const f4*)src)[2 * j + 1];
        v8h hh;
#pragma unroll
        for (int e = 0; e < 4; ++e) { hh[e] = (_Float16)f0[e]; hh[4 + e] = (_Float16)f1[e]; }
        *(v8h*)&sA[row][c0 + 8 * j] = hh;
      }
    }
    // --- stage B: 3 x (64 n-rows x 32 k), f32 -> f16 (4 threads per row)
    {
      const int row = tid >> 2;
      const int c0  = (tid & 3) * 8;
#pragma unroll
      for (int w = 0; w < 3; ++w) {
        const float* src = Wp[w] + (size_t)(n0 + row) * C_ + kk + c0;
        f4 f0 = ((const f4*)src)[0];
        f4 f1 = ((const f4*)src)[1];
        v8h hh;
#pragma unroll
        for (int e = 0; e < 4; ++e) { hh[e] = (_Float16)f0[e]; hh[4 + e] = (_Float16)f1[e]; }
        *(v8h*)&sB[w][row][c0] = hh;
      }
    }
    __syncthreads();

    const v16h a = frag_a(&sA[wave * 16][0], 40);
#pragma unroll
    for (int nt = 0; nt < 4; ++nt) {
      accq[nt] = wmma16(a, frag_b(&sB[0][nt * 16][0], 40), accq[nt]);
      acck[nt] = wmma16(a, frag_b(&sB[1][nt * 16][0], 40), acck[nt]);
      accv[nt] = wmma16(a, frag_b(&sB[2][nt * 16][0], 40), accv[nt]);
    }
    __syncthreads();
  }

  // --- store to [B,H,S,D] f16 (Q pre-scaled)
  const int n_l = lane & 15;
  const int rb  = (lane >> 4) << 3;
#pragma unroll
  for (int nt = 0; nt < 4; ++nt) {
    const int ng = n0 + nt * 16 + n_l;
    const int h  = ng >> 6;       // /64
    const int d  = ng & 63;
#pragma unroll
    for (int r = 0; r < 8; ++r) {
      const int mg = m0 + wave * 16 + rb + r;
      const int bb = mg >> 12;    // /4096
      const int ss = mg & 4095;
      const size_t o = (((size_t)bb * H_ + h) * S_ + ss) * D_ + d;
      Qh[o] = (_Float16)(accq[nt][r] * 0.125f);
      Kh[o] = (_Float16)acck[nt][r];
      Vh[o] = (_Float16)accv[nt][r];
    }
  }
}

// ---------------------------------------------------------------------------
// Kernel 2: flash attention per (b, h, 64 q-rows).  Block 128 thr = 4 waves.
// K tile staged by the Tensor Data Mover (padded rows); V staged transposed
// manually; online softmax; O accumulated in WMMA f32 fragments.
// ---------------------------------------------------------------------------
__global__ __launch_bounds__(128) void k_attn(
    const _Float16* __restrict__ Qh, const _Float16* __restrict__ Kh,
    const _Float16* __restrict__ Vh, _Float16* __restrict__ AOh)
{
  __shared__ _Float16 sK[64][72];      // [key][d]   (row pad 16B for TDM/banks)
  __shared__ _Float16 sVt[64][72];     // [d][key]   (transposed V)
  __shared__ float    sS[4][16][68];   // per-wave score tile 16x64
  __shared__ _Float16 sP[4][16][72];   // per-wave probs tile 16x64 (f16)
  __shared__ float    stats[4][16];    // per-wave per-row alpha / 1/l

  const int tid  = threadIdx.x;
  const int wave = tid >> 5;
  const int lane = tid & 31;
  const int q0   = blockIdx.x * 64;
  const int h    = blockIdx.y;
  const int b    = blockIdx.z;
  const size_t headBase = ((size_t)b * H_ + h) * (size_t)S_ * D_;

  const _Float16* gQ = Qh + headBase + (size_t)(q0 + wave * 16) * D_;
  const v16h aq0 = frag_a(gQ, D_);
  const v16h aq1 = frag_a(gQ + 32, D_);

  v8f ao[4];
#pragma unroll
  for (int nt = 0; nt < 4; ++nt) ao[nt] = vzero8();
  float m_i = -3.0e38f, l_i = 0.f;

  const int row    = lane & 15;
  const int halfid = lane >> 4;
  const int rb     = halfid << 3;

  for (int kt = 0; kt < S_ / 64; ++kt) {
    const _Float16* gK = Kh + headBase + (size_t)kt * 64 * D_;
    const _Float16* gV = Vh + headBase + (size_t)kt * 64 * D_;
    if (kt + 1 < S_ / 64) {                 // prefetch next tiles into L2
      __builtin_prefetch(gK + 64 * D_, 0, 1);
      __builtin_prefetch(gV + 64 * D_, 0, 1);
    }
#if HAVE_TDM
    // --- K tile via Tensor Data Mover: 64 rows x 16 (8B units), pad 4 DW/row
    if (wave == 0) {
      tdm_load_2d(&sK[0][0], gK, /*tile_w8=*/16, /*rows=*/64, /*stride8=*/16,
                  /*pad_iv=*/4 /*16x8B*/, /*pad_amt=*/3 /*4 DW*/);
      __builtin_amdgcn_s_wait_tensorcnt(0);
    }
    // --- V staged transposed by all threads
#pragma unroll
    for (int ii = 0; ii < 4; ++ii) {
      const int i   = tid + ii * 128;
      const int key = i >> 3;
      const int d0  = (i & 7) * 8;
      v8h vv = *(const v8h*)(gV + key * D_ + d0);
#pragma unroll
      for (int j = 0; j < 8; ++j) sVt[d0 + j][key] = vv[j];
    }
#else
    // --- stage K (row-major) and V (transposed): 64x64 halfs each
#pragma unroll
    for (int ii = 0; ii < 4; ++ii) {
      const int i   = tid + ii * 128;
      const int key = i >> 3;
      const int d0  = (i & 7) * 8;
      v8h kv = *(const v8h*)(gK + key * D_ + d0);
      *(v8h*)&sK[key][d0] = kv;
      v8h vv = *(const v8h*)(gV + key * D_ + d0);
#pragma unroll
      for (int j = 0; j < 8; ++j) sVt[d0 + j][key] = vv[j];
    }
#endif
    __syncthreads();

    // --- scores: S = Q K^T  (16 x 64 per wave)
#pragma unroll
    for (int n8 = 0; n8 < 4; ++n8) {
      v8f acc = vzero8();
      acc = wmma16(aq0, frag_b(&sK[n8 * 16][0], 72), acc);
      acc = wmma16(aq1, frag_b(&sK[n8 * 16][32], 72), acc);
      const int col = n8 * 16 + (lane & 15);
#pragma unroll
      for (int r = 0; r < 8; ++r) sS[wave][rb + r][col] = acc[r];
    }

    // --- online softmax: lane pair (l, l+16) owns row l&15 (32 cols each)
    const float* srow = &sS[wave][row][halfid * 32];
    float tmax = -3.0e38f;
#pragma unroll
    for (int c = 0; c < 32; ++c) tmax = fmaxf(tmax, srow[c]);
    tmax = fmaxf(tmax, __shfl_xor(tmax, 16, 32));
    const float mnew  = fmaxf(m_i, tmax);
    const float alpha = __expf(m_i - mnew);
    _Float16* prow = &sP[wave][row][halfid * 32];
    float sum = 0.f;
#pragma unroll
    for (int c = 0; c < 32; ++c) {
      const float p = __expf(srow[c] - mnew);
      prow[c] = (_Float16)p;
      sum += p;
    }
    sum += __shfl_xor(sum, 16, 32);
    l_i = l_i * alpha + sum;
    m_i = mnew;
    if (lane < 16) stats[wave][row] = alpha;
    __syncthreads();

    // --- rescale O by alpha, then O += P V
#pragma unroll
    for (int nt = 0; nt < 4; ++nt)
#pragma unroll
      for (int r = 0; r < 8; ++r) ao[nt][r] *= stats[wave][rb + r];

#pragma unroll
    for (int ks = 0; ks < 2; ++ks) {
      const v16h ap = frag_a(&sP[wave][0][ks * 32], 72);
#pragma unroll
      for (int nt = 0; nt < 4; ++nt)
        ao[nt] = wmma16(ap, frag_b(&sVt[nt * 16][ks * 32], 72), ao[nt]);
    }
    __syncthreads();
  }

  // --- normalize and store AO[b*S+s, h*64+d] f16
  if (lane < 16) stats[wave][row] = 1.f / l_i;
  __syncthreads();
#pragma unroll
  for (int nt = 0; nt < 4; ++nt) {
    const int col = h * D_ + nt * 16 + (lane & 15);
#pragma unroll
    for (int r = 0; r < 8; ++r) {
      const int sg = q0 + wave * 16 + rb + r;
      AOh[((size_t)b * S_ + sg) * C_ + col] = (_Float16)(ao[nt][r] * stats[wave][rb + r]);
    }
  }
}

// ---------------------------------------------------------------------------
// Kernel 3: out = AO[8192,512]f16 @ Wo^T + bo  -> f32
// A tile staged by TDM (f16 copy with padded rows); weights converted in VALU.
// ---------------------------------------------------------------------------
__global__ __launch_bounds__(256) void k_oproj(
    const _Float16* __restrict__ AO, const float* __restrict__ Wo,
    const float* __restrict__ bo, float* __restrict__ out)
{
  __shared__ _Float16 sA[128][40];
  __shared__ _Float16 sB[64][40];

  const int tid  = threadIdx.x;
  const int wave = tid >> 5;
  const int lane = tid & 31;
  const int m0   = blockIdx.x * 128;
  const int n0   = blockIdx.y * 64;

  v8f acc[4];
#pragma unroll
  for (int nt = 0; nt < 4; ++nt) acc[nt] = vzero8();

  for (int kk = 0; kk < C_; kk += 32) {
#if HAVE_TDM
    // --- A tile via TDM: 128 rows x 8 (8B units), stride 128 units, pad 4 DW
    if (wave == 0) {
      tdm_load_2d(&sA[0][0], AO + (size_t)m0 * C_ + kk,
                  /*tile_w8=*/8, /*rows=*/128, /*stride8=*/128,
                  /*pad_iv=*/3 /*8x8B*/, /*pad_amt=*/3 /*4 DW*/);
      __builtin_amdgcn_s_wait_tensorcnt(0);
    }
#else
    {
      const int row = tid >> 1;
      const int c0  = (tid & 1) * 16;
      const _Float16* src = AO + (size_t)(m0 + row) * C_ + kk + c0;
      *(v8h*)&sA[row][c0]     = ((const v8h*)src)[0];
      *(v8h*)&sA[row][c0 + 8] = ((const v8h*)src)[1];
    }
#endif
    {
      const int row = tid >> 2;
      const int c0  = (tid & 3) * 8;
      const float* src = Wo + (size_t)(n0 + row) * C_ + kk + c0;
      f4 f0 = ((const f4*)src)[0];
      f4 f1 = ((const f4*)src)[1];
      v8h hh;
#pragma unroll
      for (int e = 0; e < 4; ++e) { hh[e] = (_Float16)f0[e]; hh[4 + e] = (_Float16)f1[e]; }
      *(v8h*)&sB[row][c0] = hh;
    }
    __syncthreads();

    const v16h a = frag_a(&sA[wave * 16][0], 40);
#pragma unroll
    for (int nt = 0; nt < 4; ++nt)
      acc[nt] = wmma16(a, frag_b(&sB[nt * 16][0], 40), acc[nt]);
    __syncthreads();
  }

  const int n_l = lane & 15;
  const int rb  = (lane >> 4) << 3;
#pragma unroll
  for (int nt = 0; nt < 4; ++nt) {
    const int ng = n0 + nt * 16 + n_l;
    const float bias = bo[ng];
#pragma unroll
    for (int r = 0; r < 8; ++r) {
      const int mg = m0 + wave * 16 + rb + r;
      out[(size_t)mg * C_ + ng] = acc[nt][r] + bias;
    }
  }
}

// ---------------------------------------------------------------------------
extern "C" void kernel_launch(void* const* d_in, const int* in_sizes, int n_in,
                              void* d_out, int out_size, void* d_ws, size_t ws_size,
                              hipStream_t stream) {
  const float* X  = (const float*)d_in[0];
  const float* Wq = (const float*)d_in[1];
  const float* Wk = (const float*)d_in[2];
  const float* Wv = (const float*)d_in[3];
  const float* Wo = (const float*)d_in[4];
  const float* bo = (const float*)d_in[5];
  float* out = (float*)d_out;

  const size_t qn = (size_t)M_ * C_;          // 8192*512 halfs each
  _Float16* Qh  = (_Float16*)d_ws;
  _Float16* Kh  = Qh + qn;
  _Float16* Vh  = Kh + qn;
  _Float16* AOh = Vh + qn;                     // 32 MB total

  k_qkv  <<<dim3(M_ / 128, C_ / 64),    256, 0, stream>>>(X, Wq, Wk, Wv, Qh, Kh, Vh);
  k_attn <<<dim3(S_ / 64, H_, B_),      128, 0, stream>>>(Qh, Kh, Vh, AOh);
  k_oproj<<<dim3(M_ / 128, C_ / 64),    256, 0, stream>>>(AOh, Wo, bo, out);
}
